// MultiPhaseTemporalMemory_59974923321453
// MI455X (gfx1250) — matrix operand order repaired and checked by reading, not compiled, not run on hardware
//
#include <hip/hip_runtime.h>

#define DD     512
#define PP     8
#define NKEYS  8192
#define BATCH  8
#define TM     64          // key rows per workgroup in the WMMA kernel
#define TBLOCK 256         // 8 waves (wave32)
#define NCHUNK 64
#define CH     (NKEYS / NCHUNK)   // 128 rows per partial chunk

// LDS layout (dynamic, byte offsets from dynamic-LDS base)
#define OFF_A   0                        // [TM][DD] bf16      = 65536
#define OFF_W   (TM*DD*2)                // [DD][32] bf16 slab = 32768 (reused: raw W2|Wa f32)
#define OFF_H   (OFF_W + DD*32*2)        // [TM][DD] bf16      = 65536
#define OFF_PH  (OFF_H + TM*DD*2)        // pk+ak [2][TM][PP] f32 = 4096
#define OFF_WT  (OFF_PH + 2*TM*PP*4)     // transposed W2|Wa [2][PP][DD] f32 = 32768
#define SMEM_SZ (OFF_WT + 2*PP*DD*4)     // 200704 B

typedef __attribute__((ext_vector_type(16))) __bf16 v16bf;
typedef __attribute__((ext_vector_type(8)))  float  v8f;

__device__ __forceinline__ unsigned short f2bf(float f) {
  unsigned u = __builtin_bit_cast(unsigned, f);
  u += 0x7FFFu + ((u >> 16) & 1u);      // round-to-nearest-even
  return (unsigned short)(u >> 16);
}
__device__ __forceinline__ unsigned pack_bf(float lo, float hi) {
  return (unsigned)f2bf(lo) | ((unsigned)f2bf(hi) << 16);
}
__device__ __forceinline__ float bflo(unsigned u) {
  return __builtin_bit_cast(float, u << 16);
}
__device__ __forceinline__ float bfhi(unsigned u) {
  return __builtin_bit_cast(float, u & 0xffff0000u);
}
__device__ __forceinline__ float softplus_f(float x) {
  return (x > 20.f) ? x : log1pf(expf(x));
}

// ---------------------------------------------------------------------------
// Query phasors: aq[b][p], pq[b][p].  B=8 rows -> trivial, one block.
// ---------------------------------------------------------------------------
__global__ __launch_bounds__(256) void phasor_query_kernel(
    const float* __restrict__ query,
    const float* __restrict__ W1, const float* __restrict__ b1,
    const float* __restrict__ W2, const float* __restrict__ b2,
    const float* __restrict__ Wa, const float* __restrict__ ba,
    float* __restrict__ aq, float* __restrict__ pq)
{
  __shared__ float hq[DD];
  const int tid = threadIdx.x;
  for (int b = 0; b < BATCH; ++b) {
    for (int c = tid; c < DD; c += 256) {
      float s = b1[c];
      for (int d = 0; d < DD; ++d) s += query[b*DD + d] * W1[(size_t)d*DD + c];
      hq[c] = fmaxf(s, 0.f);
    }
    __syncthreads();
    if (tid < PP) {
      float sp = b2[tid], sa = ba[tid];
      for (int d = 0; d < DD; ++d) {
        sp += hq[d]           * W2[d*PP + tid];
        sa += query[b*DD + d] * Wa[d*PP + tid];
      }
      pq[b*PP + tid] = tanhf(sp) * 3.14159265358979323846f;
      aq[b*PP + tid] = softplus_f(sa) + 0.1f;
    }
    __syncthreads();
  }
}

// ---------------------------------------------------------------------------
// Main kernel: per 64-row key tile
//   h = relu(keys @ W1 + b1)   via v_wmma_f32_16x16x32_bf16 (b128 LDS frags)
//   phases/amps (P=8)          via vectorized VALU dots from LDS
//   alignment[b,n] = sum_p ak*aq*cos(pk - pq)
// W2/Wa staged to LDS with GLOBAL_LOAD_ASYNC_TO_LDS_B128 (+ s_wait_asynccnt).
// ---------------------------------------------------------------------------
__global__ __launch_bounds__(TBLOCK) void phasor_keys_kernel(
    const float* __restrict__ keys,
    const float* __restrict__ W1, const float* __restrict__ b1,
    const float* __restrict__ W2, const float* __restrict__ b2,
    const float* __restrict__ Wa, const float* __restrict__ ba,
    const float* __restrict__ aq, const float* __restrict__ pq,
    float* __restrict__ align_out)
{
  extern __shared__ char smem[];
  unsigned short* Abuf = (unsigned short*)(smem + OFF_A);   // keys tile, bf16
  unsigned short* Wbuf = (unsigned short*)(smem + OFF_W);   // W1 K-slab, bf16 K-major
  unsigned short* Hbuf = (unsigned short*)(smem + OFF_H);   // h tile, bf16
  float* Pbuf   = (float*)(smem + OFF_PH);                  // pk [TM][PP]
  float* AmpBuf = Pbuf + TM*PP;                             // ak [TM][PP]
  float* Wraw   = (float*)(smem + OFF_W);                   // raw W2|Wa after GEMM
  float* WT     = (float*)(smem + OFF_WT);                  // [2][PP][DD] transposed

  const int tid  = threadIdx.x;
  const int bi   = blockIdx.x;
  const int b    = bi / (NKEYS / TM);
  const int tile = bi % (NKEYS / TM);
  const int row0 = tile * TM;

  // ---- stage keys tile, f32 -> bf16, row-major in LDS (b64 stores) ------
  const float4* src = (const float4*)(keys + ((size_t)b*NKEYS + row0) * DD);
  for (int i = tid; i < TM*DD/4; i += TBLOCK) {
    float4 v = src[i];
    uint2 pk2;
    pk2.x = pack_bf(v.x, v.y);
    pk2.y = pack_bf(v.z, v.w);
    *(uint2*)&((unsigned*)Abuf)[i*2] = pk2;
  }
  __syncthreads();

  // ---- GEMM1: h = keys @ W1 ---------------------------------------------
  const int lane  = tid & 31;
  const int wid   = tid >> 5;
  const int mt    = wid >> 1;                 // 0..3  -> 16-row M tile
  const int nh    = wid & 1;                  // 0..1  -> which 256-col half
  const int mrow  = mt*16 + (lane & 15);
  const int khalf = (lane >> 4) ? 8  : 0;     // A-frag K sub-offset (elements)
  const int koff  = (lane >> 4) ? 8  : 0;     // B-frag K sub-offset (elements)
  const int ncol  = lane & 15;

  v8f acc[16];
  {
    v8f z = {0.f,0.f,0.f,0.f,0.f,0.f,0.f,0.f};
    #pragma unroll
    for (int i = 0; i < 16; ++i) acc[i] = z;
  }

  for (int kb = 0; kb < DD/32; ++kb) {
    // cooperative stage of W1 K-slab [32][DD] -> Wbuf[col*32 + kk], packed
    // K-pairs so stores are b32 and fragments are contiguous 128-bit runs.
    for (int i = tid; i < 16*(DD/4); i += TBLOCK) {
      const int kk2 = (i >> 7) * 2;          // even k within slab: 0,2,..,30
      const int n   = (i & 127) * 4;
      const float4 v0 = *(const float4*)(W1 + ((size_t)(kb*32 + kk2    ))*DD + n);
      const float4 v1 = *(const float4*)(W1 + ((size_t)(kb*32 + kk2 + 1))*DD + n);
      *(unsigned*)&Wbuf[(n+0)*32 + kk2] = pack_bf(v0.x, v1.x);
      *(unsigned*)&Wbuf[(n+1)*32 + kk2] = pack_bf(v0.y, v1.y);
      *(unsigned*)&Wbuf[(n+2)*32 + kk2] = pack_bf(v0.z, v1.z);
      *(unsigned*)&Wbuf[(n+3)*32 + kk2] = pack_bf(v0.w, v1.w);
    }
    // prefetch next K-slab into cache while this one is consumed
    if (kb + 1 < DD/32) {
      const char* nx = (const char*)(W1 + (size_t)(kb+1)*32*DD);
      for (int i = tid*128; i < 32*DD*4; i += TBLOCK*128)
        __builtin_prefetch(nx + i, 0, 0);
    }
    __syncthreads();

    // A fragment: two contiguous 16B runs per lane -> ds_load_b128 x2
    union { uint4 q[2]; v16bf v; } afrag;
    const unsigned short* arow = Abuf + mrow*DD + kb*32;
    afrag.q[0] = *(const uint4*)(arow + khalf);
    afrag.q[1] = *(const uint4*)(arow + 16 + khalf);

    #pragma unroll
    for (int nt = 0; nt < 16; ++nt) {
      const int col = (nh*16 + nt)*16 + ncol;
      union { uint4 q[2]; v16bf v; } bfrag;
      const unsigned short* wcol = Wbuf + col*32 + koff;
      bfrag.q[0] = *(const uint4*)(wcol);
      bfrag.q[1] = *(const uint4*)(wcol + 16);
      acc[nt] = __builtin_amdgcn_wmma_f32_16x16x32_bf16(
          false, afrag.v, false, bfrag.v, (short)0, acc[nt], false, false);
    }
    __syncthreads();
  }

  // ---- epilogue: bias + relu -> bf16 h in LDS ---------------------------
  const int rbase = mt*16 + 8*(lane >> 4);
  #pragma unroll
  for (int nt = 0; nt < 16; ++nt) {
    const int col = (nh*16 + nt)*16 + ncol;
    const float bias = b1[col];
    #pragma unroll
    for (int r = 0; r < 8; ++r) {
      float h = acc[nt][r] + bias;
      Hbuf[(rbase + r)*DD + col] = f2bf(fmaxf(h, 0.f));
    }
  }

  // ---- async-stage raw W2|Wa (f32, 16KB each) into freed W slab ---------
  {
    const unsigned ldsb = __builtin_amdgcn_groupstaticsize();  // dyn-LDS base
    const unsigned dst0 = ldsb + (unsigned)OFF_W;
    const unsigned long long s2 = (unsigned long long)(uintptr_t)W2;
    const unsigned long long sa = (unsigned long long)(uintptr_t)Wa;
    for (int i = tid*16; i < DD*PP*4; i += TBLOCK*16) {
      unsigned d2 = dst0 + (unsigned)i;
      unsigned da = dst0 + (unsigned)(DD*PP*4) + (unsigned)i;
      unsigned long long g2 = s2 + (unsigned)i;
      unsigned long long ga = sa + (unsigned)i;
      asm volatile("global_load_async_to_lds_b128 %0, %1, off"
                   :: "v"(d2), "v"(g2) : "memory");
      asm volatile("global_load_async_to_lds_b128 %0, %1, off"
                   :: "v"(da), "v"(ga) : "memory");
    }
    asm volatile("s_wait_asynccnt 0x0" ::: "memory");
  }
  __syncthreads();

  // transpose W2|Wa to [P][D] so the dot loop reads float4 runs
  for (int i = tid; i < DD*PP; i += TBLOCK) {
    const int d = i >> 3, p = i & 7;
    WT[p*DD + d]         = Wraw[i];             // W2^T
    WT[PP*DD + p*DD + d] = Wraw[DD*PP + i];     // Wa^T
  }
  __syncthreads();

  // ---- P=8 projections: phases/amps per (row,p), vectorized -------------
  for (int t = tid; t < TM*PP; t += TBLOCK) {
    const int row = t >> 3, p = t & 7;
    const uint4*  hr  = (const uint4*)(Hbuf + row*DD);
    const uint4*  ar  = (const uint4*)(Abuf + row*DD);
    const float4* w2r = (const float4*)(WT + p*DD);
    const float4* war = (const float4*)(WT + PP*DD + p*DD);
    float sp = b2[p], sa = ba[p];
    for (int d8 = 0; d8 < DD/8; ++d8) {
      const uint4  h4 = hr[d8];
      const uint4  a4 = ar[d8];
      const float4 w0 = w2r[2*d8], w1 = w2r[2*d8 + 1];
      const float4 x0 = war[2*d8], x1 = war[2*d8 + 1];
      sp += bflo(h4.x)*w0.x + bfhi(h4.x)*w0.y + bflo(h4.y)*w0.z + bfhi(h4.y)*w0.w
          + bflo(h4.z)*w1.x + bfhi(h4.z)*w1.y + bflo(h4.w)*w1.z + bfhi(h4.w)*w1.w;
      sa += bflo(a4.x)*x0.x + bfhi(a4.x)*x0.y + bflo(a4.y)*x0.z + bfhi(a4.y)*x0.w
          + bflo(a4.z)*x1.x + bfhi(a4.z)*x1.y + bflo(a4.w)*x1.z + bfhi(a4.w)*x1.w;
    }
    Pbuf[t]   = tanhf(sp) * 3.14159265358979323846f;
    AmpBuf[t] = softplus_f(sa) + 0.1f;
  }
  __syncthreads();

  // ---- alignment score per row ------------------------------------------
  if (tid < TM) {
    float s = 0.f;
    #pragma unroll
    for (int p = 0; p < PP; ++p)
      s += AmpBuf[tid*PP + p] * aq[b*PP + p] * cosf(Pbuf[tid*PP + p] - pq[b*PP + p]);
    align_out[(size_t)b*NKEYS + row0 + tid] = s;
  }
}

// ---------------------------------------------------------------------------
// Softmax over n (one block per batch row)
// ---------------------------------------------------------------------------
__global__ __launch_bounds__(256) void softmax_kernel(
    const float* __restrict__ align_in, float* __restrict__ w_out)
{
  __shared__ float red[256];
  const int b = blockIdx.x, tid = threadIdx.x;
  const float* x = align_in + (size_t)b*NKEYS;
  float* w = w_out + (size_t)b*NKEYS;

  float m = -3.4e38f;
  for (int i = tid; i < NKEYS; i += 256) m = fmaxf(m, x[i]);
  red[tid] = m; __syncthreads();
  for (int s = 128; s > 0; s >>= 1) {
    if (tid < s) red[tid] = fmaxf(red[tid], red[tid+s]);
    __syncthreads();
  }
  m = red[0]; __syncthreads();

  float sum = 0.f;
  for (int i = tid; i < NKEYS; i += 256) { float e = expf(x[i] - m); w[i] = e; sum += e; }
  red[tid] = sum; __syncthreads();
  for (int s = 128; s > 0; s >>= 1) {
    if (tid < s) red[tid] += red[tid+s];
    __syncthreads();
  }
  const float inv = 1.f / red[0];
  for (int i = tid; i < NKEYS; i += 256) w[i] *= inv;
}

// ---------------------------------------------------------------------------
// retrieved = weights @ values : split-n partials, b128 streaming loads.
// 128 threads x float4 = 512 cols; deterministic (no float atomics).
// ---------------------------------------------------------------------------
__global__ __launch_bounds__(128) void weighted_partial_kernel(
    const float* __restrict__ values, const float* __restrict__ w,
    float* __restrict__ partial)
{
  const int bi = blockIdx.x;
  const int b = bi >> 6, chunk = bi & 63;
  const int n0 = chunk * CH;
  const int d0 = threadIdx.x * 4;
  const float4* vp = (const float4*)(values + ((size_t)b*NKEYS + n0)*DD + d0);
  float4 accv = {0.f, 0.f, 0.f, 0.f};
  for (int r = 0; r < CH; ++r) {
    if (r + 8 < CH)
      __builtin_prefetch(vp + (size_t)(r+8)*(DD/4), 0, 0);
    const float  wr = w[(size_t)b*NKEYS + n0 + r];
    const float4 v  = vp[(size_t)r*(DD/4)];
    accv.x += wr * v.x;  accv.y += wr * v.y;
    accv.z += wr * v.z;  accv.w += wr * v.w;
  }
  *(float4*)(partial + (((size_t)b*NCHUNK) + chunk)*DD + d0) = accv;
}

__global__ __launch_bounds__(256) void reduce_out_kernel(
    const float* __restrict__ partial, float* __restrict__ out)
{
  const int b = blockIdx.x;
  for (int d = threadIdx.x; d < DD; d += 256) {
    float s = 0.f;
    for (int c = 0; c < NCHUNK; ++c)
      s += partial[(((size_t)b*NCHUNK) + c)*DD + d];
    out[(size_t)b*DD + d] = s;
  }
}

// ---------------------------------------------------------------------------
extern "C" void kernel_launch(void* const* d_in, const int* in_sizes, int n_in,
                              void* d_out, int out_size, void* d_ws, size_t ws_size,
                              hipStream_t stream) {
  const float* keys   = (const float*)d_in[0];
  const float* values = (const float*)d_in[1];
  const float* query  = (const float*)d_in[2];
  const float* W1     = (const float*)d_in[3];
  const float* b1     = (const float*)d_in[4];
  const float* W2     = (const float*)d_in[5];
  const float* b2     = (const float*)d_in[6];
  const float* Wa     = (const float*)d_in[7];
  const float* ba     = (const float*)d_in[8];
  float* out = (float*)d_out;

  float* ws      = (float*)d_ws;
  float* aq      = ws;                       // 64
  float* pq      = ws + 64;                  // 64
  float* align_  = ws + 128;                 // B*n = 65536
  float* wts     = ws + 128 + BATCH*NKEYS;   // 65536
  float* partial = ws + 128 + 2*BATCH*NKEYS; // B*NCHUNK*D = 262144

  (void)hipFuncSetAttribute((const void*)phasor_keys_kernel,
                            hipFuncAttributeMaxDynamicSharedMemorySize,
                            (int)SMEM_SZ);

  phasor_query_kernel<<<1, 256, 0, stream>>>(query, W1, b1, W2, b2, Wa, ba, aq, pq);
  phasor_keys_kernel<<<BATCH*NKEYS/TM, TBLOCK, SMEM_SZ, stream>>>(
      keys, W1, b1, W2, b2, Wa, ba, aq, pq, align_);
  softmax_kernel<<<BATCH, 256, 0, stream>>>(align_, wts);
  weighted_partial_kernel<<<BATCH*NCHUNK, 128, 0, stream>>>(values, wts, partial);
  reduce_out_kernel<<<BATCH, 256, 0, stream>>>(partial, out);
}
